// PairWeightedAveraging_49744311222649
// MI455X (gfx1250) — matrix operand order repaired and checked by reading, not compiled, not run on hardware
//
#include <hip/hip_runtime.h>
#include <hip/hip_bf16.h>

typedef __attribute__((ext_vector_type(16))) _Float16 v16h;
typedef __attribute__((ext_vector_type(8)))  _Float16 v8h;
typedef __attribute__((ext_vector_type(8)))  float    v8f;

#define S_DIM 512
#define N_DIM 384
#define CM    64
#define CZ    128
#define H_DIM 8
#define D_DIM 32
#define HD    256
#define MN_LD 72    // LDS row stride (halves) for mn tile: 144B, chunks stay 16B aligned
#define OT_LD 48    // LDS row stride (halves) for per-head gated-o tile: 96B

// ---------------- WMMA helpers -------------------------------------------

__device__ __forceinline__ v8f wmma_f16(v16h a, v16h b, v8f c) {
  // D = A(16x32 f16) * B(32x16 f16) + C(16x16 f32)
  return __builtin_amdgcn_wmma_f32_16x16x32_f16(false, a, false, b, (short)0, c,
                                                false, false);
}

// A fragment (16x32 f16, row-major source, row = lane&15).
// Per ISA 7.12.2: lanes 0-15 hold K {0..7,16..23}, lanes 16-31 hold K {8..15,24..31}.
__device__ __forceinline__ v16h load_frag_a(const _Float16* base, int ld, int k0, int lane) {
  const int r = lane & 15, hi = lane >> 4;
  const _Float16* p = base + (size_t)r * ld + k0 + 8 * hi;
  v8h lo = *(const v8h*)p;
  v8h hh = *(const v8h*)(p + 16);
  v16h o;
#pragma unroll
  for (int e = 0; e < 8; ++e) { o[e] = lo[e]; o[e + 8] = hh[e]; }
  return o;
}

// B fragment (32x16 f16), source laid out [n][k] (k contiguous), col n = lane&15.
// Lanes 0-15 hold K {0..7,16..23}, lanes 16-31 hold K {8..15,24..31}.
__device__ __forceinline__ v16h load_frag_b(const _Float16* base, int ld, int k0, int lane) {
  const int n = lane & 15, hi = lane >> 4;
  const _Float16* p = base + (size_t)n * ld + k0 + 8 * hi;
  v8h lo = *(const v8h*)p;
  v8h hh = *(const v8h*)(p + 16);
  v16h o;
#pragma unroll
  for (int e = 0; e < 8; ++e) { o[e] = lo[e]; o[e + 8] = hh[e]; }
  return o;
}

// Async copy of a 24KB contiguous region (32x384 f16) global -> LDS,
// spread over 128 threads: 12 x b128 per lane, tracked by ASYNCcnt.
__device__ __forceinline__ void async_fill_v(const _Float16* gsrc, _Float16* lds_dst,
                                             int tid) {
  const unsigned lbase = (unsigned)(size_t)lds_dst;
  const unsigned long long gbase = (unsigned long long)(size_t)gsrc;
#pragma unroll
  for (int k = 0; k < 12; ++k) {
    const unsigned off = (unsigned)(k * 128 + tid) * 16u;
    unsigned l = lbase + off;
    unsigned long long g = gbase + off;
    asm volatile("global_load_async_to_lds_b128 %0, %1, off"
                 :: "v"(l), "v"(g) : "memory");
  }
}

// LayerNorm 16 rows of Cm=64 by one wave; result f16 into LDS [16][MN_LD].
__device__ __forceinline__ void wave_ln16(const float* __restrict__ rows,
                                          const float* __restrict__ w,
                                          const float* __restrict__ b,
                                          _Float16* mn, int lane) {
  const int r = lane & 15, hi = lane >> 4;
  const float* p = rows + (size_t)r * CM + hi * 32;
  float x[32];
  float s1 = 0.f, s2 = 0.f;
#pragma unroll
  for (int c = 0; c < 32; ++c) { x[c] = p[c]; s1 += x[c]; s2 += x[c] * x[c]; }
  s1 += __shfl_xor(s1, 16, 32);
  s2 += __shfl_xor(s2, 16, 32);
  const float mu  = s1 * (1.f / 64.f);
  const float inv = rsqrtf(s2 * (1.f / 64.f) - mu * mu + 1e-5f);
#pragma unroll
  for (int c = 0; c < 32; ++c) {
    const int cc = hi * 32 + c;
    mn[r * MN_LD + cc] = (_Float16)((x[c] - mu) * inv * w[cc] + b[cc]);
  }
}

// ---------------- K0: cast weights to f16, transposed [n][k] --------------

__global__ __launch_bounds__(256) void cvt_weights_kernel(
    const float* __restrict__ Wm, const float* __restrict__ Wg,
    const float* __restrict__ Wo,
    _Float16* __restrict__ Wm_t, _Float16* __restrict__ Wg_t,
    _Float16* __restrict__ Wo_t) {
  const int t = blockIdx.x * 256 + threadIdx.x;   // 16384 threads exactly
  { // Wm/Wg: (64,256) -> [n=256][k=64]
    const int k = t >> 8, n = t & 255;
    Wm_t[n * CM + k] = (_Float16)Wm[t];
    Wg_t[n * CM + k] = (_Float16)Wg[t];
  }
  { // Wo: (256,64) -> [n=64][k=256]
    const int k = t >> 6, n = t & 63;
    Wo_t[n * HD + k] = (_Float16)Wo[t];
  }
}

// ---------------- K1: zn layernorm + pair-bias logits ---------------------

__global__ __launch_bounds__(256) void zn_logits_kernel(
    const float* __restrict__ z, const float* __restrict__ zw,
    const float* __restrict__ zb, const float* __restrict__ Wz,
    const float* __restrict__ mask, float* __restrict__ logits) {
  __shared__ float sWz[CZ * H_DIM];
  __shared__ float szw[CZ], szb[CZ];
  for (int t = threadIdx.x; t < CZ * H_DIM; t += 256) sWz[t] = Wz[t];
  for (int t = threadIdx.x; t < CZ; t += 256) { szw[t] = zw[t]; szb[t] = zb[t]; }
  __syncthreads();

  const int idx = blockIdx.x * 256 + threadIdx.x;  // 147456 threads exactly
  const int i = idx / N_DIM, j = idx - i * N_DIM;
  const float4* z4 = (const float4*)(z + (size_t)idx * CZ);

  float s1 = 0.f, s2 = 0.f;
#pragma unroll
  for (int c4 = 0; c4 < CZ / 4; ++c4) {
    float4 v = z4[c4];
    s1 += v.x + v.y + v.z + v.w;
    s2 += v.x * v.x + v.y * v.y + v.z * v.z + v.w * v.w;
  }
  const float mu  = s1 * (1.f / CZ);
  const float inv = rsqrtf(s2 * (1.f / CZ) - mu * mu + 1e-5f);

  float acc[H_DIM];
#pragma unroll
  for (int h = 0; h < H_DIM; ++h) acc[h] = 0.f;
#pragma unroll 4
  for (int c4 = 0; c4 < CZ / 4; ++c4) {
    float4 v = z4[c4];   // hot in L0
    float xn[4] = {v.x, v.y, v.z, v.w};
#pragma unroll
    for (int q = 0; q < 4; ++q) {
      const int c = c4 * 4 + q;
      const float y = (xn[q] - mu) * inv * szw[c] + szb[c];
#pragma unroll
      for (int h = 0; h < H_DIM; ++h) acc[h] += y * sWz[c * H_DIM + h];
    }
  }
  const float mj = mask[j];
#pragma unroll
  for (int h = 0; h < H_DIM; ++h)
    logits[((size_t)h * N_DIM + i) * N_DIM + j] = acc[h] + mj;
}

// ---------------- K2: softmax over j, one wave per (h,i) row --------------

__global__ __launch_bounds__(128) void softmax_kernel(
    const float* __restrict__ logits, _Float16* __restrict__ wout) {
  const int lane = threadIdx.x & 31;
  const int row  = blockIdx.x * 4 + (threadIdx.x >> 5);   // 3072 rows
  const float* lr = logits + (size_t)row * N_DIM;
  float v[12];
  float mx = -3.4e38f;
#pragma unroll
  for (int t = 0; t < 12; ++t) { v[t] = lr[lane + 32 * t]; mx = fmaxf(mx, v[t]); }
#pragma unroll
  for (int o = 16; o > 0; o >>= 1) mx = fmaxf(mx, __shfl_xor(mx, o, 32));
  float sum = 0.f;
#pragma unroll
  for (int t = 0; t < 12; ++t) { v[t] = __expf(v[t] - mx); sum += v[t]; }
#pragma unroll
  for (int o = 16; o > 0; o >>= 1) sum += __shfl_xor(sum, o, 32);
  const float rs = 1.f / sum;
  _Float16* wr = wout + (size_t)row * N_DIM;
#pragma unroll
  for (int t = 0; t < 12; ++t) wr[lane + 32 * t] = (_Float16)(v[t] * rs);
}

// ---------------- K3: mn layernorm + v = mn@Wm, stored [s][h][d][j] -------

__global__ __launch_bounds__(128) void project_v_kernel(
    const float* __restrict__ m, const float* __restrict__ mw,
    const float* __restrict__ mb, const _Float16* __restrict__ Wm_t,
    _Float16* __restrict__ vt) {
  __shared__ _Float16 mnbuf[4][16 * MN_LD];
  const int wave = threadIdx.x >> 5, lane = threadIdx.x & 31;
  const int s  = blockIdx.y;
  const int j0 = (blockIdx.x * 4 + wave) * 16;
  _Float16* mn = mnbuf[wave];

  wave_ln16(m + ((size_t)s * N_DIM + j0) * CM, mw, mb, mn, lane);
  const v16h a0 = load_frag_a(mn, MN_LD, 0, lane);
  const v16h a1 = load_frag_a(mn, MN_LD, 32, lane);
  const int hi = lane >> 4;

#pragma unroll
  for (int nt = 0; nt < HD / 16; ++nt) {          // 16 col tiles
    v8f c = {};
    const _Float16* wb = Wm_t + (size_t)(nt * 16) * CM;
    c = wmma_f16(a0, load_frag_b(wb, CM, 0, lane), c);
    c = wmma_f16(a1, load_frag_b(wb, CM, 32, lane), c);
    // D frag: row j = j0 + r + 8*hi, col hd = nt*16 + (lane&15)
    const int col = nt * 16 + (lane & 15);
    const int h = col >> 5, d = col & 31;
    v8h pk;
#pragma unroll
    for (int r = 0; r < 8; ++r) pk[r] = (_Float16)c[r];
    *(v8h*)(vt + (((size_t)s * H_DIM + h) * D_DIM + d) * N_DIM + j0 + 8 * hi) = pk;
  }
}

// ---------------- K4: fused attention + gating + output projection --------
// v[s,h] tiles (24KB) are double-buffered through LDS with async copies;
// per-head gated o is transposed through a small LDS tile and immediately
// folded into the Wo projection accumulators.

__global__ __launch_bounds__(128) void attn_out_kernel(
    const float* __restrict__ m, const float* __restrict__ mw,
    const float* __restrict__ mb, const _Float16* __restrict__ wsm,
    const _Float16* __restrict__ vt, const _Float16* __restrict__ Wg_t,
    const _Float16* __restrict__ Wo_t, float* __restrict__ out) {
  __shared__ _Float16 vstage[2][D_DIM * N_DIM];       // 2 x 24KB
  __shared__ _Float16 mnbuf[4][16 * MN_LD];           // 9KB
  __shared__ _Float16 otile[4][16 * OT_LD];           // 6KB
  const int tid  = threadIdx.x;
  const int wave = tid >> 5, lane = tid & 31;
  const int s  = blockIdx.y;
  const int i0 = (blockIdx.x * 4 + wave) * 16;
  const int r = lane & 15, hi = lane >> 4;
  _Float16* mn = mnbuf[wave];
  _Float16* ot = otile[wave];

  const _Float16* vsrc = vt + (size_t)s * H_DIM * D_DIM * N_DIM;
  async_fill_v(vsrc, vstage[0], tid);                 // prefetch head 0

  wave_ln16(m + ((size_t)s * N_DIM + i0) * CM, mw, mb, mn, lane);
  const v16h amn0 = load_frag_a(mn, MN_LD, 0, lane);
  const v16h amn1 = load_frag_a(mn, MN_LD, 32, lane);

  v8f cf[4] = {};                                     // output accumulators
  for (int h = 0; h < H_DIM; ++h) {
    if (h < H_DIM - 1) {
      async_fill_v(vsrc + (size_t)(h + 1) * D_DIM * N_DIM, vstage[(h + 1) & 1], tid);
      asm volatile("s_wait_asynccnt 0xc" ::: "memory");   // head h's 12 ops done
    } else {
      asm volatile("s_wait_asynccnt 0x0" ::: "memory");
    }
    __syncthreads();                                  // all quarters visible
    const _Float16* vb = vstage[h & 1];

    // o[i-tile, d] = w[h, i-tile, :] @ v[s, :, h, :]
    v8f co0 = {}, co1 = {};
    const _Float16* wbase = wsm + ((size_t)h * N_DIM + i0) * N_DIM;
#pragma unroll 4
    for (int kt = 0; kt < N_DIM / 32; ++kt) {         // 12 K-steps over j
      const v16h aw = load_frag_a(wbase, N_DIM, kt * 32, lane);
      co0 = wmma_f16(aw, load_frag_b(vb,              N_DIM, kt * 32, lane), co0);
      co1 = wmma_f16(aw, load_frag_b(vb + 16 * N_DIM, N_DIM, kt * 32, lane), co1);
    }

    // gates: sigmoid(mn @ Wg[:, h*32 .. h*32+31])
    v8f cg0 = {}, cg1 = {};
    const _Float16* gb = Wg_t + (size_t)(h * 32) * CM;
    cg0 = wmma_f16(amn0, load_frag_b(gb,           CM, 0,  lane), cg0);
    cg0 = wmma_f16(amn1, load_frag_b(gb,           CM, 32, lane), cg0);
    cg1 = wmma_f16(amn0, load_frag_b(gb + 16 * CM, CM, 0,  lane), cg1);
    cg1 = wmma_f16(amn1, load_frag_b(gb + 16 * CM, CM, 32, lane), cg1);

    // gate in the D-fragment layout and transpose via the small LDS tile
#pragma unroll
    for (int rr = 0; rr < 8; ++rr) {
      const int row = rr + 8 * hi;
      const float g0 = 1.f / (1.f + __expf(-cg0[rr]));
      const float g1 = 1.f / (1.f + __expf(-cg1[rr]));
      ot[row * OT_LD + r]      = (_Float16)(co0[rr] * g0);
      ot[row * OT_LD + 16 + r] = (_Float16)(co1[rr] * g1);
    }

    // fold this head into the Wo projection: cf += o_h[16,32] @ Wo[h*32.., :]
    const v16h ao = load_frag_a(ot, OT_LD, 0, lane);
#pragma unroll
    for (int nt = 0; nt < 4; ++nt)
      cf[nt] = wmma_f16(ao, load_frag_b(Wo_t + (size_t)(nt * 16) * HD, HD, h * 32, lane),
                        cf[nt]);
    __syncthreads();                                  // done reading vstage[h&1]
  }

#pragma unroll
  for (int nt = 0; nt < 4; ++nt) {
    float* op = out + ((size_t)s * N_DIM + i0) * CM + nt * 16 + r;
#pragma unroll
    for (int rr = 0; rr < 8; ++rr) op[(size_t)(rr + 8 * hi) * CM] = cf[nt][rr];
  }
}

// ---------------- host launch ---------------------------------------------

extern "C" void kernel_launch(void* const* d_in, const int* in_sizes, int n_in,
                              void* d_out, int out_size, void* d_ws, size_t ws_size,
                              hipStream_t stream) {
  const float* m    = (const float*)d_in[0];
  const float* z    = (const float*)d_in[1];
  const float* mask = (const float*)d_in[2];
  const float* mnw  = (const float*)d_in[3];
  const float* mnb  = (const float*)d_in[4];
  const float* znw  = (const float*)d_in[5];
  const float* znb  = (const float*)d_in[6];
  const float* Wm   = (const float*)d_in[7];
  const float* Wg   = (const float*)d_in[8];
  const float* Wz   = (const float*)d_in[9];
  const float* Wo   = (const float*)d_in[10];
  float* out = (float*)d_out;

  char* ws = (char*)d_ws;
  size_t off = 0;
  auto take = [&](size_t bytes) -> char* {
    char* p = ws + off;
    off = (off + bytes + 255) & ~(size_t)255;
    return p;
  };
  float*    logits = (float*)   take((size_t)H_DIM * N_DIM * N_DIM * 4);
  _Float16* wsm    = (_Float16*)take((size_t)H_DIM * N_DIM * N_DIM * 2);
  _Float16* vt     = (_Float16*)take((size_t)S_DIM * H_DIM * D_DIM * N_DIM * 2);
  _Float16* Wm_t   = (_Float16*)take((size_t)HD * CM * 2);
  _Float16* Wg_t   = (_Float16*)take((size_t)HD * CM * 2);
  _Float16* Wo_t   = (_Float16*)take((size_t)CM * HD * 2);

  cvt_weights_kernel<<<dim3(64), dim3(256), 0, stream>>>(Wm, Wg, Wo, Wm_t, Wg_t, Wo_t);
  zn_logits_kernel<<<dim3(576), dim3(256), 0, stream>>>(z, znw, znb, Wz, mask, logits);
  softmax_kernel<<<dim3(768), dim3(128), 0, stream>>>(logits, wsm);
  project_v_kernel<<<dim3(6, 512), dim3(128), 0, stream>>>(m, mnw, mnb, Wm_t, vt);
  attn_out_kernel<<<dim3(6, 512), dim3(128), 0, stream>>>(m, mnw, mnb, wsm, vt, Wg_t, Wo_t, out);
}
